// AE_RNN_32083405701121
// MI455X (gfx1250) — matrix operand order repaired and checked by reading, not compiled, not run on hardware
//
#include <hip/hip_runtime.h>
#include <hip/hip_fp16.h>

// ---------------------------------------------------------------------------
// Problem constants (match reference)
// ---------------------------------------------------------------------------
#define BB   256
#define UDIM 16
#define YDIM 16
#define ZDIM 32
#define HDIM 128
#define LNUM 2
#define TT   2048

typedef __attribute__((ext_vector_type(16))) _Float16 v16h;
typedef __attribute__((ext_vector_type(8)))  float    v8f;

// ---------------------------------------------------------------------------
// WMMA fragment helpers (CDNA5 ISA 7.12.2 layouts, wave32)
// A (16x32 f16): lane&15 = row M; lanes<16 hold K 0-7 / 16-23, lanes>=16 hold
//                K 8-15 / 24-31 (element j -> K kb+j, element 8+j -> K 16+kb+j)
// B (32x16 f16) as W^T (torch Linear): lane&15 = col N; lanes<16 carry K 0-7 / 16-23,
//                lanes>=16 carry K 8-15 / 24-31 (same order as A)
// C/D (16x16 f32): VGPR i, lanes<16 -> M=i, lanes>=16 -> M=8+i; N = lane&15
// ---------------------------------------------------------------------------
__device__ __forceinline__ v16h ld_a(const _Float16* __restrict__ X, int ld, int lane) {
  const int r  = lane & 15;
  const int kb = (lane >> 4) << 3;
  const _Float16* p = X + (size_t)r * ld;
  v16h a;
#pragma unroll
  for (int j = 0; j < 8; ++j) {
    a[j]     = p[kb + j];
    a[8 + j] = p[16 + kb + j];
  }
  return a;
}

__device__ __forceinline__ v16h ld_bt(const _Float16* __restrict__ W, int ld,
                                      int n0, int k0, int lane) {
  const int n  = n0 + (lane & 15);
  const int kb = k0 + ((lane >> 4) << 3);
  const _Float16* p = W + (size_t)n * ld + kb;
  v16h b;
#pragma unroll
  for (int e = 0; e < 8; ++e) { b[e] = p[e]; b[8 + e] = p[16 + e]; }
  return b;
}

__device__ __forceinline__ v8f wmma16(v16h a, v16h b, v8f c) {
  return __builtin_amdgcn_wmma_f32_16x16x32_f16(false, a, false, b,
                                                (short)0, c, false, false);
}

__device__ __forceinline__ float sigmoidf_(float x) {
  return 1.0f / (1.0f + __expf(-x));
}

// CDNA5 async global->LDS copy (ASYNCcnt-tracked, ISA 15.18.3 op 98).
// VDST = VGPR with wave-relative LDS byte offset, VADDR = 64-bit global addr.
__device__ __forceinline__ void async_g2l_b128(unsigned lds_off, const void* g) {
  asm volatile("global_load_async_to_lds_b128 %0, %1, off"
               :: "v"(lds_off), "v"(g) : "memory");
}
__device__ __forceinline__ void wait_asynccnt0() {
  asm volatile("s_wait_asynccnt 0" ::: "memory");
}
__device__ __forceinline__ unsigned lds_off_of(const void* p) {
  // Generic LDS pointers carry the wave-relative LDS offset in the low 32 bits.
  // NOTE: must only be used as a runtime value (never in a static initializer).
  return (unsigned)(uintptr_t)p;
}

// ---------------------------------------------------------------------------
// fp32 -> fp16 weight conversion
// ---------------------------------------------------------------------------
__global__ __launch_bounds__(256) void k_cvt(const float* __restrict__ s,
                                             _Float16* __restrict__ d, int n) {
  int i = blockIdx.x * blockDim.x + threadIdx.x;
  if (i < n) d[i] = (_Float16)s[i];
}

// ---------------------------------------------------------------------------
// Kernel 1: phi_u MLP over all (t, batch-tile). Grid (16, T), 256 threads.
// pu[t,b,:] = relu(u_t @ W1^T + b1) @ W2^T + b2   (stored fp16)
// ---------------------------------------------------------------------------
__global__ __launch_bounds__(256) void k_phi_u(
    const float* __restrict__ u, const _Float16* __restrict__ w1,
    const float* __restrict__ b1, const _Float16* __restrict__ w2,
    const float* __restrict__ b2, _Float16* __restrict__ pu) {
  __shared__ _Float16 xin[16 * 32];    // u_t tile, K padded 16 -> 32
  __shared__ _Float16 w1p[128 * 32];   // W1 padded in K
  __shared__ _Float16 hmid[16 * 128];  // hidden after relu

  const int t    = blockIdx.y;
  const int mb   = blockIdx.x;
  const int tid  = threadIdx.x;
  const int wv   = tid >> 5;
  const int lane = tid & 31;

  for (int i = tid; i < 16 * 32; i += 256) {
    int r = i >> 5, c = i & 31;
    float v = 0.0f;
    if (c < UDIM)  // u is (B, UD, T): gather the t-slice
      v = u[((size_t)(mb * 16 + r) * UDIM + c) * TT + t];
    xin[i] = (_Float16)v;
  }
  for (int i = tid; i < 128 * 32; i += 256) {
    int r = i >> 5, c = i & 31;
    w1p[i] = (c < UDIM) ? w1[r * UDIM + c] : (_Float16)0.0f;
  }
  __syncthreads();

  const int n0    = wv * 16;
  const int n     = n0 + (lane & 15);
  const int mbase = (lane >> 4) << 3;

  {  // layer 1 (K = 32 padded, single WMMA per wave)
    v8f acc = {};
    acc = wmma16(ld_a(xin, 32, lane), ld_bt(w1p, 32, n0, 0, lane), acc);
    const float bias = b1[n];
#pragma unroll
    for (int i = 0; i < 8; ++i) {
      float v = acc[i] + bias;
      hmid[(mbase + i) * 128 + n] = (_Float16)(v > 0.0f ? v : 0.0f);
    }
  }
  __syncthreads();
  {  // layer 2 (K = 128)
    v8f acc = {};
#pragma unroll
    for (int k0 = 0; k0 < 128; k0 += 32)
      acc = wmma16(ld_a(hmid + k0, 128, lane), ld_bt(w2, 128, n0, k0, lane), acc);
    const float bias = b2[n];
    _Float16* out = pu + ((size_t)t * BB + mb * 16) * HDIM;
#pragma unroll
    for (int i = 0; i < 8; ++i)
      out[(size_t)(mbase + i) * HDIM + n] = (_Float16)(acc[i] + bias);
  }
}

// ---------------------------------------------------------------------------
// Kernel 2: sequential 2-layer GRU scan. ONE persistent workgroup, 32 waves.
// LDS (exactly 320KB, the CDNA5 per-WG cap):
//   3 x 64KB rotating h-state buffers (halves [0, 98304))
//   2 x 64KB double-buffered pu panels (halves [98304, 163840)), filled by
//     global_load_async_to_lds_b128 (ASYNCcnt) one timestep ahead of use.
// All LDS addresses are computed as runtime scalars off `smem` (no pointer
// aggregates -> no addrspacecast static initializers for ld.lld to choke on).
// ---------------------------------------------------------------------------
__global__ __launch_bounds__(1024) void k_scan(
    const float* __restrict__ h0, const _Float16* __restrict__ pu,
    const _Float16* __restrict__ wih, const _Float16* __restrict__ whh,
    _Float16* __restrict__ hhist) {
  extern __shared__ _Float16 smem[];  // 5 * 32768 halves = 320 KB
  const int tid  = threadIdx.x;
  const int wv   = tid >> 5;
  const int lane = tid & 31;

  const unsigned puLdsBase = lds_off_of(smem) + 196608u;  // byte offset of panel 0

  // kick off async staging of panel 0 (each thread copies 4 x 16B)
#pragma unroll
  for (int j = 0; j < 4; ++j) {
    const unsigned boff = (unsigned)((j * 1024 + tid) * 16);
    async_g2l_b128(puLdsBase + boff, (const char*)pu + boff);
  }

  for (int i = tid; i < 32768; i += 1024) {
    smem[i]         = (_Float16)h0[i];          // layer-0 state (buffer 0)
    smem[32768 + i] = (_Float16)h0[32768 + i];  // layer-1 state (buffer 1)
  }
  __syncthreads();

  int cur0 = 0, cur1 = 1, fre = 2;

#pragma unroll 1
  for (int t = 0; t < TT; ++t) {
    wait_asynccnt0();   // panel t landed (this wave's issues)
    __syncthreads();    // ... and visible to the whole workgroup
    if (t + 1 < TT) {   // stage panel t+1 while we compute on panel t
      const char* gsrc = (const char*)(pu + (size_t)(t + 1) * 32768);
      const unsigned dst = puLdsBase + (unsigned)((t + 1) & 1) * 65536u;
#pragma unroll
      for (int j = 0; j < 4; ++j) {
        const unsigned boff = (unsigned)((j * 1024 + tid) * 16);
        async_g2l_b128(dst + boff, gsrc + boff);
      }
    }
    // record pre-update h[-1] for the loss branch (b128 granularity)
    {
      const float4* s4 = (const float4*)(smem + (size_t)cur1 * 32768);
      float4* d4 = (float4*)(hhist + (size_t)t * 32768);
      for (int i = tid; i < 4096; i += 1024) d4[i] = s4[i];
    }
    const _Float16* put = smem + 98304 + (size_t)(t & 1) * 32768;

#pragma unroll 1
    for (int layer = 0; layer < LNUM; ++layer) {
      const _Float16* xin  = (layer == 0) ? put : (smem + (size_t)fre * 32768);
      const _Float16* hst  = smem + (size_t)(layer == 0 ? cur0 : cur1) * 32768;
      _Float16*       hout = smem + (size_t)(layer == 0 ? fre : cur0) * 32768;
      const _Float16* Wi = wih + (size_t)layer * 3 * HDIM * HDIM;
      const _Float16* Wh = whh + (size_t)layer * 3 * HDIM * HDIM;

#pragma unroll 1
      for (int jj = 0; jj < 4; ++jj) {  // 128 tile-jobs over 32 waves
        const int job = wv + (jj << 5);
        const int m0  = (job >> 3) << 4;  // batch-tile
        const int n0  = (job & 7) << 4;   // H-column tile
        v8f ir = {}, iz = {}, in_ = {}, hr = {}, hz = {}, hn = {};
#pragma unroll
        for (int k0 = 0; k0 < 128; k0 += 32) {
          v16h aI = ld_a(xin + (size_t)m0 * 128 + k0, 128, lane);
          v16h aH = ld_a(hst + (size_t)m0 * 128 + k0, 128, lane);
          ir  = wmma16(aI, ld_bt(Wi, 128, n0,       k0, lane), ir);
          iz  = wmma16(aI, ld_bt(Wi, 128, 128 + n0, k0, lane), iz);
          in_ = wmma16(aI, ld_bt(Wi, 128, 256 + n0, k0, lane), in_);
          hr  = wmma16(aH, ld_bt(Wh, 128, n0,       k0, lane), hr);
          hz  = wmma16(aH, ld_bt(Wh, 128, 128 + n0, k0, lane), hz);
          hn  = wmma16(aH, ld_bt(Wh, 128, 256 + n0, k0, lane), hn);
        }
        const int n     = n0 + (lane & 15);
        const int mbase = m0 + ((lane >> 4) << 3);
#pragma unroll
        for (int i = 0; i < 8; ++i) {
          float r  = sigmoidf_(ir[i] + hr[i]);
          float z  = sigmoidf_(iz[i] + hz[i]);
          float nn = tanhf(in_[i] + r * hn[i]);
          float ho = (float)hst[(size_t)(mbase + i) * 128 + n];
          hout[(size_t)(mbase + i) * 128 + n] =
              (_Float16)((1.0f - z) * nn + z * ho);
        }
      }
      __syncthreads();
    }
    // rotate roles: (cur0, cur1, fre) <- (fre, cur0, cur1)
    int nc0 = fre, nc1 = cur0, nf = cur1;
    cur0 = nc0; cur1 = nc1; fre = nf;
  }
}

// ---------------------------------------------------------------------------
// Kernel 3: time-parallel loss branch. Grid (16, T), 256 threads.
// d = MLP([pu_t | h[-1]]); x = d@Wx^T+b; yh = MLP(MLP(x)); loss += |yh-y_t|^2
// ---------------------------------------------------------------------------
__global__ __launch_bounds__(256) void k_loss(
    const _Float16* __restrict__ pu, const _Float16* __restrict__ hhist,
    const float* __restrict__ y,
    const _Float16* __restrict__ wd1, const float* __restrict__ bd1,
    const _Float16* __restrict__ wd2, const float* __restrict__ bd2,
    const _Float16* __restrict__ wxm, const float* __restrict__ bxm,
    const _Float16* __restrict__ wpx1, const float* __restrict__ bpx1,
    const _Float16* __restrict__ wpx2, const float* __restrict__ bpx2,
    const _Float16* __restrict__ wm1, const float* __restrict__ bm1,
    const _Float16* __restrict__ wm2, const float* __restrict__ bm2,
    float* __restrict__ out) {
  __shared__ _Float16 bufA[16 * 128];
  __shared__ _Float16 bufB[16 * 128];
  __shared__ _Float16 xlds[16 * 32];

  const int t    = blockIdx.y;
  const int mb   = blockIdx.x;
  const int tid  = threadIdx.x;
  const int wv   = tid >> 5;
  const int lane = tid & 31;
  const int n0    = wv * 16;
  const int n     = n0 + (lane & 15);
  const int mbase = (lane >> 4) << 3;

  const _Float16* put = pu    + ((size_t)t * BB + mb * 16) * HDIM;
  const _Float16* hht = hhist + ((size_t)t * BB + mb * 16) * HDIM;

  // dynn layer 1: K = 256 over concat [pu | h[-1]], relu
  {
    v8f acc = {};
#pragma unroll
    for (int k0 = 0; k0 < 256; k0 += 32) {
      const _Float16* src = (k0 < 128) ? (put + k0) : (hht + (k0 - 128));
      acc = wmma16(ld_a(src, 128, lane), ld_bt(wd1, 256, n0, k0, lane), acc);
    }
    const float bias = bd1[n];
#pragma unroll
    for (int i = 0; i < 8; ++i) {
      float v = acc[i] + bias;
      bufA[(mbase + i) * 128 + n] = (_Float16)(v > 0.0f ? v : 0.0f);
    }
  }
  __syncthreads();
  // dynn layer 2 -> d
  {
    v8f acc = {};
#pragma unroll
    for (int k0 = 0; k0 < 128; k0 += 32)
      acc = wmma16(ld_a(bufA + k0, 128, lane), ld_bt(wd2, 128, n0, k0, lane), acc);
    const float bias = bd2[n];
#pragma unroll
    for (int i = 0; i < 8; ++i)
      bufB[(mbase + i) * 128 + n] = (_Float16)(acc[i] + bias);
  }
  __syncthreads();
  // x = d @ x_mean^T + b  (ZD = 32 -> waves 0,1 only)
  if (wv < 2) {
    v8f acc = {};
#pragma unroll
    for (int k0 = 0; k0 < 128; k0 += 32)
      acc = wmma16(ld_a(bufB + k0, 128, lane), ld_bt(wxm, 128, n0, k0, lane), acc);
    const float bias = bxm[n];
#pragma unroll
    for (int i = 0; i < 8; ++i)
      xlds[(mbase + i) * 32 + n] = (_Float16)(acc[i] + bias);
  }
  __syncthreads();
  // phi_x layer 1: K = 32, relu
  {
    v8f acc = {};
    acc = wmma16(ld_a(xlds, 32, lane), ld_bt(wpx1, 32, n0, 0, lane), acc);
    const float bias = bpx1[n];
#pragma unroll
    for (int i = 0; i < 8; ++i) {
      float v = acc[i] + bias;
      bufA[(mbase + i) * 128 + n] = (_Float16)(v > 0.0f ? v : 0.0f);
    }
  }
  __syncthreads();
  // phi_x layer 2 -> px
  {
    v8f acc = {};
#pragma unroll
    for (int k0 = 0; k0 < 128; k0 += 32)
      acc = wmma16(ld_a(bufA + k0, 128, lane), ld_bt(wpx2, 128, n0, k0, lane), acc);
    const float bias = bpx2[n];
#pragma unroll
    for (int i = 0; i < 8; ++i)
      bufB[(mbase + i) * 128 + n] = (_Float16)(acc[i] + bias);
  }
  __syncthreads();
  // menn layer 1: relu
  {
    v8f acc = {};
#pragma unroll
    for (int k0 = 0; k0 < 128; k0 += 32)
      acc = wmma16(ld_a(bufB + k0, 128, lane), ld_bt(wm1, 128, n0, k0, lane), acc);
    const float bias = bm1[n];
#pragma unroll
    for (int i = 0; i < 8; ++i) {
      float v = acc[i] + bias;
      bufA[(mbase + i) * 128 + n] = (_Float16)(v > 0.0f ? v : 0.0f);
    }
  }
  __syncthreads();
  // menn layer 2 (YD = 16 -> wave 0), squared-error reduction
  if (wv == 0) {
    v8f acc = {};
#pragma unroll
    for (int k0 = 0; k0 < 128; k0 += 32)
      acc = wmma16(ld_a(bufA + k0, 128, lane), ld_bt(wm2, 128, 0, k0, lane), acc);
    const int nn   = lane & 15;
    const float bias = bm2[nn];
    float part = 0.0f;
#pragma unroll
    for (int i = 0; i < 8; ++i) {
      int row  = mb * 16 + mbase + i;  // y is (B, YD, T)
      float yv = y[((size_t)row * YDIM + nn) * TT + t];
      float d  = acc[i] + bias - yv;
      part += d * d;
    }
#pragma unroll
    for (int off = 16; off > 0; off >>= 1)
      part += __shfl_xor(part, off, 32);
    if (lane == 0) atomicAdd(out, part);
  }
}

// ---------------------------------------------------------------------------
// Host launcher
// ---------------------------------------------------------------------------
extern "C" void kernel_launch(void* const* d_in, const int* in_sizes, int n_in,
                              void* d_out, int out_size, void* d_ws, size_t ws_size,
                              hipStream_t stream) {
  (void)in_sizes; (void)n_in; (void)ws_size;

  const float* u        = (const float*)d_in[0];
  const float* y        = (const float*)d_in[1];
  const float* h0       = (const float*)d_in[2];
  const float* phi_u_w1 = (const float*)d_in[3];
  const float* phi_u_b1 = (const float*)d_in[4];
  const float* phi_u_w2 = (const float*)d_in[5];
  const float* phi_u_b2 = (const float*)d_in[6];
  const float* dynn_w1  = (const float*)d_in[7];
  const float* dynn_b1  = (const float*)d_in[8];
  const float* dynn_w2  = (const float*)d_in[9];
  const float* dynn_b2  = (const float*)d_in[10];
  const float* x_mean_w = (const float*)d_in[11];
  const float* x_mean_b = (const float*)d_in[12];
  const float* phi_x_w1 = (const float*)d_in[13];
  const float* phi_x_b1 = (const float*)d_in[14];
  const float* phi_x_w2 = (const float*)d_in[15];
  const float* phi_x_b2 = (const float*)d_in[16];
  const float* menn_w1  = (const float*)d_in[17];
  const float* menn_b1  = (const float*)d_in[18];
  const float* menn_w2  = (const float*)d_in[19];
  const float* menn_b2  = (const float*)d_in[20];
  const float* gru_wih  = (const float*)d_in[21];
  const float* gru_whh  = (const float*)d_in[22];

  // Workspace layout (fp16 elements)
  const size_t NPU = (size_t)TT * BB * HDIM;  // 67,108,864
  _Float16* ws   = (_Float16*)d_ws;
  _Float16* puW  = ws;
  _Float16* hhW  = ws + NPU;
  _Float16* wp   = hhW + NPU;
  _Float16* f_pu_w1 = wp; wp += HDIM * UDIM;
  _Float16* f_pu_w2 = wp; wp += HDIM * HDIM;
  _Float16* f_d_w1  = wp; wp += HDIM * 2 * HDIM;
  _Float16* f_d_w2  = wp; wp += HDIM * HDIM;
  _Float16* f_xm    = wp; wp += ZDIM * HDIM;
  _Float16* f_px1   = wp; wp += HDIM * ZDIM;
  _Float16* f_px2   = wp; wp += HDIM * HDIM;
  _Float16* f_m1    = wp; wp += HDIM * HDIM;
  _Float16* f_m2    = wp; wp += YDIM * HDIM;
  _Float16* f_wih   = wp; wp += (size_t)LNUM * 3 * HDIM * HDIM;
  _Float16* f_whh   = wp; wp += (size_t)LNUM * 3 * HDIM * HDIM;

  (void)hipMemsetAsync(d_out, 0, sizeof(float) * (size_t)out_size, stream);

  auto cvt = [&](const float* s, _Float16* d, int n) {
    k_cvt<<<(n + 255) / 256, 256, 0, stream>>>(s, d, n);
  };
  cvt(phi_u_w1, f_pu_w1, HDIM * UDIM);
  cvt(phi_u_w2, f_pu_w2, HDIM * HDIM);
  cvt(dynn_w1,  f_d_w1,  HDIM * 2 * HDIM);
  cvt(dynn_w2,  f_d_w2,  HDIM * HDIM);
  cvt(x_mean_w, f_xm,    ZDIM * HDIM);
  cvt(phi_x_w1, f_px1,   HDIM * ZDIM);
  cvt(phi_x_w2, f_px2,   HDIM * HDIM);
  cvt(menn_w1,  f_m1,    HDIM * HDIM);
  cvt(menn_w2,  f_m2,    YDIM * HDIM);
  cvt(gru_wih,  f_wih,   LNUM * 3 * HDIM * HDIM);
  cvt(gru_whh,  f_whh,   LNUM * 3 * HDIM * HDIM);

  // 1) time-parallel phi_u
  k_phi_u<<<dim3(BB / 16, TT), 256, 0, stream>>>(u, f_pu_w1, phi_u_b1,
                                                 f_pu_w2, phi_u_b2, puW);

  // 2) sequential GRU scan: one persistent 32-wave workgroup, 320KB LDS
  //    (3 h-state buffers + 2 async-staged pu panels)
  const int scanLds = 5 * BB * HDIM * (int)sizeof(_Float16);  // 327680
  (void)hipFuncSetAttribute(reinterpret_cast<const void*>(k_scan),
                            hipFuncAttributeMaxDynamicSharedMemorySize, scanLds);
  k_scan<<<1, 1024, scanLds, stream>>>(h0, puW, f_wih, f_whh, hhW);

  // 3) time-parallel loss branch
  k_loss<<<dim3(BB / 16, TT), 256, 0, stream>>>(
      puW, hhW, y,
      f_d_w1, dynn_b1, f_d_w2, dynn_b2,
      f_xm, x_mean_b,
      f_px1, phi_x_b1, f_px2, phi_x_b2,
      f_m1, menn_b1, f_m2, menn_b2,
      (float*)d_out);
}
